// SpectralKoopmanSDE_49873160241878
// MI455X (gfx1250) — compile-verified, matrix operand order checked by reading, not run on hardware
//
#include <hip/hip_runtime.h>
#include <hip/hip_bf16.h>
#include <math.h>

// ---------------------------------------------------------------------------
// SpectralKoopmanSDE on MI455X (gfx1250, wave32, WMMA f16 16x16x32 / f32 acc)
//
// lam depends only on the W-frequency bin, so the H-axis FFT in rfft2/irfft2
// cancels: the spectral step is a per-row circular convolution along W with
// k = irfft(lam) (128 real taps per (b,c,r)).
//   Per (b,c,r):  Y[128x128] = X[128x128] @ Circulant(k)   -> WMMA GEMM
// Convs are implicit-GEMM over NHWC-f16 padded activations; the per-block
// activation halo tile is staged into LDS by the Tensor Data Mover (TDM).
// ---------------------------------------------------------------------------

typedef __attribute__((ext_vector_type(16))) _Float16 v16h;
typedef __attribute__((ext_vector_type(8)))  float    v8f;
typedef __attribute__((ext_vector_type(4)))  unsigned u32x4;
typedef __attribute__((ext_vector_type(8)))  int      i32x8;
typedef __attribute__((ext_vector_type(4)))  int      i32x4;

union Frag { u32x4 q[2]; v16h v; _Float16 h[16]; };
union H2   { _Float16 h[2]; unsigned u; };

#define BB   8
#define CC   64
#define HH   128
#define WW   128
#define RR   4
#define WFQ  65
#define PADN 130

// ---- workspace layout (bytes) ----
static constexpr size_t SPBYTES      = (size_t)BB * PADN * PADN * CC * 2; // padded NHWC f16
static constexpr size_t XPAD_OFF     = 0;                      // conv1 input, later aliased as conv2 output
static constexpr size_t FEAT1_OFF    = XPAD_OFF + SPBYTES;     // conv1 output
static constexpr size_t WFBYTES      = (size_t)9 * 2 * 4 * 32 * 8 * 4;   // swizzled weights (dwords)
static constexpr size_t WF1_OFF      = FEAT1_OFF + SPBYTES;
static constexpr size_t WF2_OFF      = WF1_OFF + WFBYTES;
static constexpr size_t COLSUM_OFF   = WF2_OFF + WFBYTES;
static constexpr size_t COLSUM_BYTES = (size_t)BB * WW * CC * 4;
static constexpr size_t G_OFF        = COLSUM_OFF + COLSUM_BYTES;
static constexpr size_t G_BYTES      = (size_t)BB * CC * 4;
static constexpr size_t LAM_BYTES    = (size_t)BB * CC * RR * WFQ * 4;
static constexpr size_t LAMR_OFF     = G_OFF + G_BYTES;
static constexpr size_t LAMI_OFF     = LAMR_OFF + LAM_BYTES;
static constexpr size_t KC_OFF       = LAMI_OFF + LAM_BYTES;   // 8*64*4*128 f32 = 1 MB

// ---------------------------------------------------------------------------
// TDM helper: 2D tile load, data_size = 2 bytes (f16 elements).
//   global tile:  tile_d1 rows x tile_d0 elements, row stride = stride0 elems
//   packed contiguously into LDS at lds_byte_off.
// Descriptor layout per CDNA5 ISA ch.8 (D# groups 0/1; groups 2/3 zero = 2D).
__device__ __forceinline__ void tdm_load_2d_f16(unsigned lds_byte_off,
                                                const void* gaddr,
                                                unsigned tile_d0, unsigned tile_d1,
                                                unsigned stride0) {
    u32x4 g0 = {};
    g0[0] = 1u;                                    // count=1, user descriptor
    g0[1] = lds_byte_off;                          // lds_addr [63:32]
    size_t ga = (size_t)gaddr;
    g0[2] = (unsigned)ga;                          // global_addr low  [95:64]
    g0[3] = (unsigned)((ga >> 32) & 0x1FFFFFFu)    // global_addr high [120:96]
          | (2u << 30);                            // type = 2 ("image")
    i32x8 g1 = {};
    g1[0] = (int)(1u << 16);                       // data_size = 1 -> 2 bytes
    g1[1] = (int)(tile_d0 << 16);                  // tensor_dim0[15:0] in [31:16]
    g1[2] = (int)((tile_d0 >> 16) | (tile_d1 << 16)); // tdim0 hi | tensor_dim1 lo
    g1[3] = (int)((tile_d1 >> 16) | (tile_d0 << 16)); // tdim1 hi | tile_dim0
    g1[4] = (int)tile_d1;                          // tile_dim1 (tile_dim2 = 0)
    g1[5] = (int)stride0;                          // tensor_dim0_stride low32
    g1[6] = 0;                                     // stride hi | dim1_stride lo
    g1[7] = 0;
    i32x4 gz = {};
#if defined(__clang_major__) && __clang_major__ >= 23
    i32x8 gz8 = {};
    __builtin_amdgcn_tensor_load_to_lds(g0, g1, gz, gz, gz8, 0);
#else
    __builtin_amdgcn_tensor_load_to_lds(g0, g1, gz, gz, 0);
#endif
}

// ---------------------------------------------------------------------------
__global__ void k_zero(unsigned* p, int n) {
    int i = blockIdx.x * 256 + threadIdx.x;
    if (i < n) p[i] = 0u;
}

// x_curr [B,C,H,W] f32 -> padded NHWC f16 [B][130][130][C], LDS tile transpose.
// Block = (b, h, 32-wide w tile): coalesced 128B row reads, contiguous-c writes.
__global__ __launch_bounds__(256) void k_packx(const float* __restrict__ x,
                                               _Float16* __restrict__ xp) {
    __shared__ _Float16 tr[32][72];                 // [w][c], padded rows
    int blk = blockIdx.x;
    int wt = blk & 3, h = (blk >> 2) & 127, b = blk >> 9;
    int w0 = wt * 32;
    int tid = threadIdx.x;
    int wlo = tid & 31, c0 = (tid >> 5) * 8;
    const float* xr = x + (((size_t)b * CC) * HH + h) * WW + w0 + wlo;
    for (int j = 0; j < 8; ++j)
        tr[wlo][c0 + j] = (_Float16)xr[(size_t)(c0 + j) * HH * WW];
    __syncthreads();
    int c = tid & 63;
    int wl = tid >> 6;                              // 0..3
    _Float16* op = xp + ((size_t)(b * PADN + h + 1) * PADN + (w0 + 1)) * CC + c;
    for (int j = 0; j < 8; ++j)
        op[(size_t)(wl * 8 + j) * CC] = tr[wl * 8 + j][c];
}

// conv weights OIHW f32 -> per-lane swizzled B-fragments (f16x2 dwords)
// layout: [tap t 0..9)[kchunk kc 0..2)[ntile nt 0..4)[lane 0..32)[dword j 0..8)
__global__ void k_packw(const float* __restrict__ wgt, unsigned* __restrict__ dst) {
    int u = blockIdx.x * 256 + threadIdx.x;
    if (u >= 2304) return;
    int lane = u & 31, nt = (u >> 5) & 3, kc = (u >> 7) & 1, t = u >> 8;
    int ky = t / 3, kx = t % 3;
    int n = nt * 16 + (lane & 15);
    int klo = (lane >> 4) * 8;
    for (int j = 0; j < 8; ++j) {
        int kk = klo + ((j < 4) ? 2 * j : 16 + 2 * (j - 4));
        int c0 = kc * 32 + kk;
        H2 p;
        p.h[0] = (_Float16)wgt[(((size_t)n * CC + c0) * 3 + ky) * 3 + kx];
        p.h[1] = (_Float16)wgt[(((size_t)n * CC + c0 + 1) * 3 + ky) * 3 + kx];
        dst[(size_t)u * 8 + j] = p.u;
    }
}

// Implicit-GEMM 3x3 SAME conv + bias + SiLU.  One wave = 16 pixels x 16 outch.
// Block activation halo (3 rows x 34 px x 64 ch f16) staged to LDS by the TDM;
// all 8 waves (4 nt x 2 pixel tiles) share it. 18 WMMAs per wave.
__global__ __launch_bounds__(256) void k_conv(const _Float16* __restrict__ src,
                                              const unsigned* __restrict__ wf,
                                              const float* __restrict__ bias,
                                              _Float16* __restrict__ dst) {
    __shared__ __align__(16) _Float16 atile[3 * 34 * CC];   // 13056 B
    int blk = blockIdx.x;
    int hB = (blk >> 2) & 127;          // block-uniform output row
    int bB = blk >> 9;                  // block-uniform batch
    int wb0 = (2 * blk) & 7;            // first of the 2 pixel tiles in block
    int s0 = wb0 * 16;                  // first staged padded pixel

    if (threadIdx.x < 32) {             // wave 0 drives the TDM
        const void* ga = (const void*)(src + ((size_t)(bB * PADN + hB) * PADN + s0) * CC);
        tdm_load_2d_f16((unsigned)(size_t)(void*)atile, ga,
                        /*tile_d0=*/34 * CC, /*tile_d1=*/3,
                        /*stride0=*/PADN * CC);
        __builtin_amdgcn_s_wait_tensorcnt(0);
    }
    __syncthreads();

    int wave = blk * 8 + (threadIdx.x >> 5);
    int lane = threadIdx.x & 31;
    int nt = wave & 3, wb = (wave >> 2) & 7;
    int ln = lane & 15, hi = lane >> 4, klo = hi * 8;

    v8f acc = {};
    for (int t = 0; t < 9; ++t) {
        int dy = t / 3 - 1, dx = t % 3 - 1;
        int pix = (wb - wb0) * 16 + ln + 1 + dx;          // 0..33
        int abase = (dy + 1) * (34 * CC) + pix * CC;
        for (int kc = 0; kc < 2; ++kc) {
            Frag a, w;
            a.q[0] = *(const u32x4*)&atile[abase + kc * 32 + klo];
            a.q[1] = *(const u32x4*)&atile[abase + kc * 32 + klo + 16];
            const u32x4* bp =
                (const u32x4*)(wf + (size_t)((((t * 2 + kc) * 4 + nt) * 32 + lane)) * 8);
            w.q[0] = bp[0];
            w.q[1] = bp[1];
            acc = __builtin_amdgcn_wmma_f32_16x16x32_f16(false, a.v, false, w.v,
                                                         (short)0, acc, false, false);
        }
    }
    int n = nt * 16 + ln;
    float bn = bias[n];
    _Float16* ob = dst + (size_t)(bB * PADN + hB + 1) * PADN * CC;
    for (int i = 0; i < 8; ++i) {
        int m = i + hi * 8;
        float v = acc[i] + bn;
        v = v / (1.f + __expf(-v));                        // SiLU
        ob[(size_t)(wb * 16 + m + 1) * CC + n] = (_Float16)v;
    }
}

// colsum[b][w][c] = sum over h of feat2[b,h,w,c]
__global__ void k_colsum(const _Float16* __restrict__ feat, float* __restrict__ colsum) {
    int b = blockIdx.x >> 7, w = blockIdx.x & 127, c = threadIdx.x;
    const _Float16* p = feat + ((size_t)(b * PADN + 1) * PADN + (w + 1)) * CC + c;
    float s = 0.f;
    for (int h = 0; h < 128; ++h) s += (float)p[(size_t)h * PADN * CC];
    colsum[((size_t)b * 128 + w) * 64 + c] = s;
}

// g[b][c] = sigmoid(mean(feat2) @ fc_w.T + fc_b)
__global__ void k_g(const float* __restrict__ colsum, const float* __restrict__ fcw,
                    const float* __restrict__ fcb, float* __restrict__ g) {
    __shared__ float mean[64];
    int b = blockIdx.x, c = threadIdx.x;
    float m = 0.f;
    for (int w = 0; w < 128; ++w) m += colsum[((size_t)b * 128 + w) * 64 + c];
    m *= (1.f / 16384.f);
    mean[c] = m;
    __syncthreads();
    float a = fcb[c];
    for (int j = 0; j < 64; ++j) a += mean[j] * fcw[c * 64 + j];
    g[b * 64 + c] = 1.f / (1.f + __expf(-a));
}

// token -> head -> nu/theta -> lam[b][c][r][wf]
__global__ __launch_bounds__(256) void k_headlam(const float* __restrict__ colsum,
                                                 const float* __restrict__ g,
                                                 const float* __restrict__ hw,
                                                 const float* __restrict__ hb,
                                                 const float* __restrict__ dt,
                                                 float* __restrict__ lamR,
                                                 float* __restrict__ lamI) {
    __shared__ float tok[64];
    int b = blockIdx.x / WFQ, wf = blockIdx.x % WFQ;
    int tid = threadIdx.x;
    if (tid < 64) {
        int s = (wf * 128) / WFQ;
        int e = ((wf + 1) * 128 + WFQ - 1) / WFQ;   // ceil (adaptive pool bins)
        float a = 0.f;
        for (int w = s; w < e; ++w) a += colsum[((size_t)b * 128 + w) * 64 + tid];
        tok[tid] = (a * (1.f / 128.f)) * g[b * 64 + tid] / (float)(e - s);
    }
    __syncthreads();
    int c = tid >> 2, r = tid & 3;
    int row = c * 12 + r * 3;
    float p0 = hb[row], p1 = hb[row + 1];
    const float* w0 = hw + (size_t)row * 64;
    for (int j = 0; j < 64; ++j) {
        float tk = tok[j];
        p0 += tk * w0[j];
        p1 += tk * w0[64 + j];
    }
    float nu_rate = (p0 > 20.f) ? p0 : log1pf(__expf(p0));        // softplus
    float th_rate = tanhf(p1) * 3.14159265358979323846f;
    float d = dt[b];
    float mag = __expf(-nu_rate * d);
    float sv, cv;
    __sincosf(th_rate * d, &sv, &cv);
    size_t idx = (((size_t)b * 64 + c) * 4 + r) * WFQ + wf;
    lamR[idx] = mag * cv;
    lamI[idx] = mag * sv;
}

// k = irfft(lam) : 128 real taps per (b,c,r).  Twiddles via 128-entry LUT.
__global__ void k_kcirc(const float* __restrict__ lamR, const float* __restrict__ lamI,
                        float* __restrict__ kc) {
    __shared__ float Lr[WFQ], Li[WFQ], ct[128], st[128];
    int blk = blockIdx.x, t = threadIdx.x;
    if (t < WFQ) { Lr[t] = lamR[(size_t)blk * WFQ + t]; Li[t] = lamI[(size_t)blk * WFQ + t]; }
    float sv, cv;
    __sincosf((float)t * (2.f * 3.14159265358979323846f / 128.f), &sv, &cv);
    ct[t] = cv; st[t] = sv;
    __syncthreads();
    float acc = Lr[0] + ((t & 1) ? -Lr[64] : Lr[64]);
    for (int f = 1; f < 64; ++f) {
        int m = (f * t) & 127;
        acc += 2.f * (Lr[f] * ct[m] - Li[f] * st[m]);
    }
    kc[(size_t)blk * 128 + t] = acc * (1.f / 128.f);
}

// Y[h][w'] = sum_w X[h][w] * k[(w'-w) mod 128]   per (b,c,r) via WMMA.
// Only 8 distinct circulant B-fragments exist (diag = (n0-k0)/16 mod 8).
__global__ __launch_bounds__(256) void k_circ_gemm(const float* __restrict__ ht,
                                                   const float* __restrict__ kc,
                                                   float* __restrict__ out) {
    __shared__ _Float16 xt[128][136];   // +8 f16 row pad to stagger LDS banks
    __shared__ _Float16 kf[128];
    int blk = blockIdx.x;
    int b = blk >> 8, c = (blk >> 2) & 63, r = blk & 3;
    int tid = threadIdx.x;

    const float* Xg = ht + ((size_t)(b * 64 + c)) * 128 * 128 * 4 + r;
    for (int j = 0; j < 64; ++j) {
        int lin = j * 256 + tid;
        int h = lin >> 7, w = lin & 127;
        xt[h][w] = (_Float16)Xg[(size_t)(h * 128 + w) * 4];
    }
    if (tid < 128) kf[tid] = (_Float16)kc[(size_t)blk * 128 + tid];
    __syncthreads();

    int wave = tid >> 5, lane = tid & 31, ln = lane & 15, hi = lane >> 4, klo = hi * 8;
    int m0 = wave * 16;

    Frag Bf[8];
    for (int d = 0; d < 8; ++d)
        for (int j = 0; j < 8; ++j) {
            int kk = klo + ((j < 4) ? 2 * j : 16 + 2 * (j - 4));
            Bf[d].h[2 * j]     = kf[(d * 16 + ln - kk) & 127];
            Bf[d].h[2 * j + 1] = kf[(d * 16 + ln - kk - 1) & 127];
        }

    Frag Af[4];
    for (int kt = 0; kt < 4; ++kt) {
        int k0 = kt * 32;
        Af[kt].q[0] = *(const u32x4*)&xt[m0 + ln][k0 + klo];
        Af[kt].q[1] = *(const u32x4*)&xt[m0 + ln][k0 + klo + 16];
    }

    float* Og = out + ((size_t)(b * 64 + c)) * 128 * 128 * 4 + r;
    for (int nt = 0; nt < 8; ++nt) {
        v8f acc = {};
        for (int kt = 0; kt < 4; ++kt) {
            int d = (nt - 2 * kt) & 7;   // (n0 - k0)/16 mod 8, k0 = 32*kt
            acc = __builtin_amdgcn_wmma_f32_16x16x32_f16(false, Af[kt].v, false, Bf[d].v,
                                                         (short)0, acc, false, false);
        }
        int wcol = nt * 16 + ln;
        for (int i = 0; i < 8; ++i) {
            int h = m0 + i + hi * 8;
            Og[(size_t)(h * 128 + wcol) * 4] = acc[i];
        }
    }
}

// ---------------------------------------------------------------------------
extern "C" void kernel_launch(void* const* d_in, const int* in_sizes, int n_in,
                              void* d_out, int out_size, void* d_ws, size_t ws_size,
                              hipStream_t stream) {
    (void)in_sizes; (void)n_in; (void)out_size; (void)ws_size;
    const float* h_trans = (const float*)d_in[0];
    const float* x_curr  = (const float*)d_in[1];
    const float* dt      = (const float*)d_in[2];
    const float* conv1_w = (const float*)d_in[3];
    const float* conv1_b = (const float*)d_in[4];
    const float* conv2_w = (const float*)d_in[5];
    const float* conv2_b = (const float*)d_in[6];
    const float* fc_w    = (const float*)d_in[7];
    const float* fc_b    = (const float*)d_in[8];
    const float* head_w  = (const float*)d_in[9];
    const float* head_b  = (const float*)d_in[10];
    float* out = (float*)d_out;

    char* ws = (char*)d_ws;
    _Float16* xpad   = (_Float16*)(ws + XPAD_OFF);
    _Float16* feat1  = (_Float16*)(ws + FEAT1_OFF);
    _Float16* feat2  = (_Float16*)(ws + XPAD_OFF);   // alias: xpad dead after conv1
    unsigned* wf1    = (unsigned*)(ws + WF1_OFF);
    unsigned* wf2    = (unsigned*)(ws + WF2_OFF);
    float*    colsum = (float*)(ws + COLSUM_OFF);
    float*    g      = (float*)(ws + G_OFF);
    float*    lamR   = (float*)(ws + LAMR_OFF);
    float*    lamI   = (float*)(ws + LAMI_OFF);
    float*    kcirc  = (float*)(ws + KC_OFF);

    // 1. zero both padded NHWC buffers (xpad + feat1 are contiguous)
    int zn = (int)(2 * SPBYTES / 4);
    k_zero<<<(zn + 255) / 256, 256, 0, stream>>>((unsigned*)ws, zn);

    // 2. pack input & weights
    k_packx<<<BB * HH * 4, 256, 0, stream>>>(x_curr, xpad);
    k_packw<<<9, 256, 0, stream>>>(conv1_w, wf1);
    k_packw<<<9, 256, 0, stream>>>(conv2_w, wf2);

    // 3. convs (implicit GEMM, WMMA f16, TDM-staged activations, fused SiLU)
    k_conv<<<4096, 256, 0, stream>>>(xpad, wf1, conv1_b, feat1);
    k_conv<<<4096, 256, 0, stream>>>(feat1, wf2, conv2_b, feat2);

    // 4. gating + tokens + head -> lam
    k_colsum<<<BB * WW, 64, 0, stream>>>(feat2, colsum);
    k_g<<<BB, 64, 0, stream>>>(colsum, fc_w, fc_b, g);
    k_headlam<<<BB * WFQ, 256, 0, stream>>>(colsum, g, head_w, head_b, dt, lamR, lamI);

    // 5. k = irfft(lam), then per-(b,c,r) circulant GEMM (H-axis FFT cancels)
    k_kcirc<<<BB * CC * RR, 128, 0, stream>>>(lamR, lamI, kcirc);
    k_circ_gemm<<<BB * CC * RR, 256, 0, stream>>>(h_trans, kcirc, out);
}
